// CompILE_74758200754671
// MI455X (gfx1250) — matrix operand order repaired and checked by reading, not compiled, not run on hardware
//
#include <hip/hip_runtime.h>
#include <math.h>

typedef __attribute__((ext_vector_type(16))) _Float16 v16h;
typedef __attribute__((ext_vector_type(8)))  _Float16 v8h;
typedef __attribute__((ext_vector_type(8)))  float    v8f;

// Problem constants
#define BB   64
#define TT   256
#define HH   512
#define LL   64
#define VV   500
#define SS   3
#define H4   2048
#define BT   (BB*TT)

// ---------------- workspace layout (bytes) ----------------
constexpr size_t WS_WCAT = 0;                                            // [2H x 4H] f16
constexpr size_t WS_X    = WS_WCAT + sizeof(_Float16)*(size_t)2*HH*H4;   // [B*T x H] f16
constexpr size_t WS_ENCH = WS_X    + sizeof(_Float16)*(size_t)BT*HH;     // [B*T x H] f16
constexpr size_t WS_WZ1  = WS_ENCH + sizeof(_Float16)*(size_t)BT*HH;
constexpr size_t WS_WZ2  = WS_WZ1  + sizeof(_Float16)*(size_t)HH*HH;
constexpr size_t WS_WB1  = WS_WZ2  + sizeof(_Float16)*(size_t)HH*2*LL;
constexpr size_t WS_WD1  = WS_WB1  + sizeof(_Float16)*(size_t)HH*HH;
constexpr size_t WS_WD2  = WS_WD1  + sizeof(_Float16)*(size_t)LL*HH;
constexpr size_t WS_H0   = WS_WD2  + sizeof(_Float16)*(size_t)HH*512;    // padded 512x512
constexpr size_t WS_H1   = WS_H0   + sizeof(_Float16)*(size_t)BB*HH;
constexpr size_t WS_C    = WS_H1   + sizeof(_Float16)*(size_t)BB*HH;
constexpr size_t WS_MASK = WS_C    + sizeof(float)*(size_t)BB*HH;
constexpr size_t WS_LOGM = WS_MASK + sizeof(float)*(size_t)BT;
constexpr size_t WS_LGW  = WS_LOGM + sizeof(float)*(size_t)BT;
constexpr size_t WS_SB   = WS_LGW  + sizeof(float)*(size_t)BT;
constexpr size_t WS_RDH  = WS_SB   + sizeof(float)*(size_t)BT;
constexpr size_t WS_HZ   = WS_RDH  + sizeof(_Float16)*(size_t)BB*HH;
constexpr size_t WS_Z    = WS_HZ   + sizeof(_Float16)*(size_t)BB*HH;
constexpr size_t WS_HD   = WS_Z    + sizeof(_Float16)*(size_t)BB*LL;
constexpr size_t WS_PRED = WS_HD   + sizeof(_Float16)*(size_t)BB*HH;

// ---------------- output layout (floats) ----------------
constexpr size_t ENC_OFF  = 0;
constexpr size_t REC_OFF  = ENC_OFF  + (size_t)SS*BB*TT*HH;
constexpr size_t MASK_OFF = REC_OFF  + (size_t)SS*BB*TT*VV;
constexpr size_t LOGB_OFF = MASK_OFF + (size_t)(SS-1)*BB*TT;
constexpr size_t SB_OFF   = LOGB_OFF + (size_t)(SS-1)*BB*TT;
constexpr size_t LOGZ_OFF = SB_OFF   + (size_t)SS*BB*TT;
constexpr size_t SZ_OFF   = LOGZ_OFF + (size_t)SS*BB*2*LL;

__device__ __forceinline__ float sigmoidf_(float x){ return 1.0f/(1.0f+expf(-x)); }

// Fragment layout (this file's contiguous k order, k = 16*(lane/16) + e), stored so each lane's 16 halves are contiguous:
//   slot index = lane*16 + e
// A (16x32, MxK): lane = (r&15) + ((k&16)?16:0); e = k&15
// B (32x16, KxN): lane = c + ((k&16)?16:0);      e = k&15

// ---------------- small utility kernels ----------------
__global__ void cvt_kernel(const float* __restrict__ src, _Float16* __restrict__ dst, size_t n){
  size_t i = (size_t)blockIdx.x*256 + threadIdx.x;
  if (i < n) dst[i] = (_Float16)src[i];
}
__global__ void cvt_pad_kernel(const float* __restrict__ src, _Float16* __restrict__ dst){
  int i = blockIdx.x*256 + threadIdx.x;          // 512*512
  if (i >= 512*512) return;
  int r = i >> 9, c = i & 511;
  dst[i] = (c < VV) ? (_Float16)src[r*VV + c] : (_Float16)0.f;
}
__global__ void gather_kernel(const int* __restrict__ actions, const float* __restrict__ embed,
                              _Float16* __restrict__ Xh){
  int bt = blockIdx.x;
  int a  = actions[bt];
  for (int h = threadIdx.x; h < HH; h += 256)
    Xh[(size_t)bt*HH + h] = (_Float16)embed[(size_t)a*HH + h];
}
__global__ void init_hc_kernel(_Float16* h0, _Float16* h1, float* cf){
  int i = blockIdx.x*256 + threadIdx.x;          // B*H
  h0[i] = (_Float16)0.f; h1[i] = (_Float16)0.f; cf[i] = 0.f;
}
__global__ void init_mask_kernel(float* mask, float* logmask){
  int i = blockIdx.x*256 + threadIdx.x;          // B*T
  mask[i] = 1.f; logmask[i] = 0.f;
}
__global__ void fill_scalar_kernel(float* dst, const float* val, int n){
  int i = blockIdx.x*256 + threadIdx.x;
  if (i < n) dst[i] = val[0];
}

// ---------------- LSTM timestep: z = [x_t,h] @ [Wx;Wh] + b, fused gate math ----------------
// grid (H/16, 2): blockIdx.x = 16-col group of H, blockIdx.y = 32-row half of B
// Software-pipelined: regs prefetch k+64, LDS double-buffer, one barrier per K-step.
__global__ __launch_bounds__(256) void lstm_step_kernel(
    const _Float16* __restrict__ Xh,    // (B*T, H) f16
    const _Float16* __restrict__ Wcat,  // (2H, 4H) f16
    const float* __restrict__ b_lstm,   // (4H)
    const float* __restrict__ mask,     // (B, T)
    const _Float16* __restrict__ hr,    // h read  (B, H) f16
    _Float16* __restrict__ hw,          // h write (B, H) f16
    float* __restrict__ cf,             // c state (B, H) f32
    float* __restrict__ enc,            // (S,B,T,H) output base
    int seg, int t)
{
  __shared__ alignas(32) _Float16 Asf[2][2*512];    // ping-pong, fragment layout
  __shared__ alignas(32) _Float16 Bsf[2][4*512];
  __shared__ float zs[4*32*16];
  const int tid  = threadIdx.x;
  const int lane = tid & 31, wave = tid >> 5;
  const int jg   = blockIdx.x;                      // cols jg*16..+15 of H
  const int mh   = blockIdx.y;                      // rows mh*32..+31 of B
  const int g    = wave >> 1;                       // gate
  const int ml   = wave & 1;                        // local m tile

  // A staging geometry (threads 0..127, one v8h chunk each)
  const bool aAct = tid < 128;
  const int ar  = tid >> 2, ach = tid & 3;
  const int arg = mh*32 + ar;
  const int akl = ach*8;
  const int aslot = (ar >> 4)*512 + ((ar & 15) + ((akl & 16) ? 16 : 0))*16 + ((akl & 8) ? 8 : 0);
  // B staging geometry (all 256 threads, one v8h chunk each, scatter store)
  const int bg  = tid >> 6, brem = tid & 63;
  const int bkr = brem >> 1, bch = brem & 1;
  const int blnk = (bkr & 16) ? 16 : 0, bek = bkr & 15;

  auto loadA = [&](int k0) -> v8h {
    int kk = k0 + akl;
    const _Float16* src = (k0 < HH) ? (Xh + ((size_t)arg*TT + t)*HH + kk)
                                    : (hr + (size_t)arg*HH + (kk - HH));
    return *(const v8h*)src;
  };
  auto loadB = [&](int k0) -> v8h {
    const _Float16* src = Wcat + (size_t)(k0 + bkr)*H4 + bg*HH + jg*16 + bch*8;
    if (k0 + 64 < 2*HH) __builtin_prefetch(src + (size_t)64*H4, 0, 1);
    return *(const v8h*)src;
  };
  auto storeA = [&](int buf, v8h d){ *(v8h*)(&Asf[buf][aslot]) = d; };
  auto storeB = [&](int buf, v8h d){
    _Float16* dst = &Bsf[buf][bg*512 + bek];
#pragma unroll
    for (int j = 0; j < 8; ++j){ int c = bch*8 + j; dst[(c + blnk)*16] = d[j]; }
  };

  // pipeline prologue
  v8h aR{}, bR{};
  if (aAct) { aR = loadA(0); storeA(0, aR); }
  bR = loadB(0); storeB(0, bR);
  __syncthreads();
  if (aAct) aR = loadA(32);
  bR = loadB(32);

  v8f acc = {};
  int buf = 0;
  for (int k0 = 0; k0 < 2*HH; k0 += 32){
    v16h a = *(const v16h*)(&Asf[buf][ml*512 + lane*16]);
    v16h b = *(const v16h*)(&Bsf[buf][g*512  + lane*16]);
    acc = __builtin_amdgcn_wmma_f32_16x16x32_f16(false, a, false, b, (short)0, acc, false, false);
    if (k0 + 32 < 2*HH){
      if (aAct) storeA(buf ^ 1, aR);
      storeB(buf ^ 1, bR);
    }
    __syncthreads();
    if (k0 + 64 < 2*HH){
      if (aAct) aR = loadA(k0 + 64);
      bR = loadB(k0 + 64);
    }
    buf ^= 1;
  }
  // stage z tiles (+bias) for cross-wave gate combination
  {
    int col  = lane & 15;
    float bv = b_lstm[g*HH + jg*16 + col];
    int radd = (lane & 16) ? 8 : 0;
#pragma unroll
    for (int vr = 0; vr < 8; ++vr)
      zs[((size_t)g*32 + ml*16 + vr + radd)*16 + col] = acc[vr] + bv;
  }
  __syncthreads();
  // gate math + state update + enc write
  for (int i = tid; i < 32*16; i += 256){
    int r = i >> 4, cc = i & 15;
    int rg = mh*32 + r;
    int n  = jg*16 + cc;
    float iv = zs[(0*32 + r)*16 + cc];
    float fv = zs[(1*32 + r)*16 + cc];
    float gv = zs[(2*32 + r)*16 + cc];
    float ov = zs[(3*32 + r)*16 + cc];
    float co = cf[(size_t)rg*HH + n];
    float cn = sigmoidf_(fv)*co + sigmoidf_(iv)*tanhf(gv);
    float hn = sigmoidf_(ov)*tanhf(cn);
    enc[((((size_t)seg*BB + rg)*TT) + t)*HH + n] = hn;
    float m = mask[(size_t)rg*TT + t];
    hw[(size_t)rg*HH + n] = (_Float16)(m*hn);
    cf[(size_t)rg*HH + n] = m*cn;
  }
}

// ---------------- generic WMMA GEMM: out = act(A @ B + bias) ----------------
// mode 0: plain    mode 1: relu    mode 2: relu, *wvec[col], row-sum -> atomicAdd outvec[row]
__global__ __launch_bounds__(256) void gemm_wmma_kernel(
    const _Float16* __restrict__ A, int lda,     // M x K
    const _Float16* __restrict__ Bm, int ldb,    // K x N
    const float* __restrict__ bias,
    int K, int mode,
    float* __restrict__ out, int ldo, int ncols,
    _Float16* __restrict__ outh, int ldoh,
    const float* __restrict__ wvec, float* __restrict__ outvec)
{
  __shared__ alignas(32) _Float16 Asf[2][4*512];   // ping-pong, fragment layout
  __shared__ alignas(32) _Float16 Bsf[2][4*512];
  __shared__ float red[64*4];
  const int tid  = threadIdx.x;
  const int lane = tid & 31, wave = tid >> 5;
  const int row0 = blockIdx.y * 64;
  const int col0 = blockIdx.x * 64;
  const int nt   = wave >> 1;
  const int m0   = (wave & 1)*2, m1 = m0 + 1;

  const int ar  = tid >> 2, ach = tid & 3;
  const int akl = ach*8;
  const int aslot = (ar >> 4)*512 + ((ar & 15) + ((akl & 16) ? 16 : 0))*16 + ((akl & 8) ? 8 : 0);
  const int bkr = tid >> 3, bch = tid & 7;
  const int blnk = (bkr & 16) ? 16 : 0, bek = bkr & 15;

  auto loadA = [&](int k0) -> v8h {
    return *(const v8h*)(A + (size_t)(row0 + ar)*lda + k0 + akl);
  };
  auto loadB = [&](int k0) -> v8h {
    return *(const v8h*)(Bm + (size_t)(k0 + bkr)*ldb + col0 + bch*8);
  };
  auto storeA = [&](int buf, v8h d){ *(v8h*)(&Asf[buf][aslot]) = d; };
  auto storeB = [&](int buf, v8h d){
#pragma unroll
    for (int j = 0; j < 8; ++j){
      int c = bch*8 + j;
      Bsf[buf][(c >> 4)*512 + ((c & 15) + blnk)*16 + bek] = d[j];
    }
  };

  // pipeline prologue
  v8h aR = loadA(0), bR = loadB(0);
  storeA(0, aR); storeB(0, bR);
  __syncthreads();
  if (32 < K){ aR = loadA(32); bR = loadB(32); }

  v8f acc0 = {}; v8f acc1 = {};
  int buf = 0;
  for (int k0 = 0; k0 < K; k0 += 32){
    v16h bfr = *(const v16h*)(&Bsf[buf][nt*512 + lane*16]);
    v16h a0  = *(const v16h*)(&Asf[buf][m0*512 + lane*16]);
    acc0 = __builtin_amdgcn_wmma_f32_16x16x32_f16(false, a0, false, bfr, (short)0, acc0, false, false);
    v16h a1  = *(const v16h*)(&Asf[buf][m1*512 + lane*16]);
    acc1 = __builtin_amdgcn_wmma_f32_16x16x32_f16(false, a1, false, bfr, (short)0, acc1, false, false);
    if (k0 + 32 < K){ storeA(buf ^ 1, aR); storeB(buf ^ 1, bR); }
    __syncthreads();
    if (k0 + 64 < K){ aR = loadA(k0 + 64); bR = loadB(k0 + 64); }
    buf ^= 1;
  }
  const int colw = col0 + nt*16 + (lane & 15);
  const float bcol = (bias && colw < ncols) ? bias[colw] : 0.f;
  const float wv   = (mode == 2) ? wvec[colw] : 0.f;
  const int radd   = (lane & 16) ? 8 : 0;
#pragma unroll
  for (int i = 0; i < 2; ++i){
    v8f& acc = i ? acc1 : acc0;
    int m = i ? m1 : m0;
#pragma unroll
    for (int vr = 0; vr < 8; ++vr){
      int rowl = m*16 + vr + radd;
      float v = acc[vr] + bcol;
      if (mode >= 1) v = fmaxf(v, 0.f);
      if (mode == 2){
        float s = v * wv;                         // reduce across 16 lanes (same row)
        s += __shfl_xor(s, 1, 16);
        s += __shfl_xor(s, 2, 16);
        s += __shfl_xor(s, 4, 16);
        s += __shfl_xor(s, 8, 16);
        if ((lane & 15) == 0) red[rowl*4 + nt] = s;
      } else if (colw < ncols){
        int row = row0 + rowl;
        if (out)  out [(size_t)row*ldo  + colw] = v;
        if (outh) outh[(size_t)row*ldoh + colw] = (_Float16)v;
      }
    }
  }
  if (mode == 2){
    __syncthreads();
    if (tid < 64){
      float s = red[tid*4+0] + red[tid*4+1] + red[tid*4+2] + red[tid*4+3];
      atomicAdd(&outvec[row0 + tid], s);
    }
  }
}

// ---------------- heads ----------------
__global__ void softmax_gumbel_kernel(const float* __restrict__ lw, const float* __restrict__ gumbel,
                                      float* __restrict__ logb_out, float* __restrict__ sb_out,
                                      float* __restrict__ sb_ws){
  int b = blockIdx.x, t = threadIdx.x;           // 256 threads
  __shared__ float sm[256];
  float logit = (t == 0) ? -1000000000.0f : lw[b*TT + t];
  logb_out[b*TT + t] = logit;
  float x = logit + gumbel[b*TT + t];            // TEMP_B == 1
  sm[t] = x; __syncthreads();
  for (int s = 128; s > 0; s >>= 1){ if (t < s) sm[t] = fmaxf(sm[t], sm[t+s]); __syncthreads(); }
  float mx = sm[0]; __syncthreads();
  float e = expf(x - mx);
  sm[t] = e; __syncthreads();
  for (int s = 128; s > 0; s >>= 1){ if (t < s) sm[t] += sm[t+s]; __syncthreads(); }
  float p = e / sm[0];
  sb_out[b*TT + t] = p; sb_ws[b*TT + t] = p;
}
__global__ void onehot_kernel(const int* __restrict__ lengths, float* __restrict__ sb_out,
                              float* __restrict__ sb_ws){
  int b = blockIdx.x, t = threadIdx.x;
  float v = (t == lengths[b] - 1) ? 1.f : 0.f;
  sb_out[b*TT + t] = v; sb_ws[b*TT + t] = v;
}
__global__ void readout_kernel(const float* __restrict__ enc_seg, const float* __restrict__ sb,
                               _Float16* __restrict__ rh){
  int idx = blockIdx.x*256 + threadIdx.x;        // B*H
  int b = idx >> 9, h = idx & 511;
  const float* ep = enc_seg + (size_t)b*TT*HH + h;
  const float* sp = sb + (size_t)b*TT;
  float acc = 0.f;
  for (int t = 0; t < TT-1; ++t) acc += ep[(size_t)t*HH] * sp[t+1];
  rh[idx] = (_Float16)acc;
}
__global__ void samplez_kernel(const float* __restrict__ logz, const float* __restrict__ eps,
                               float* __restrict__ sz_out, _Float16* __restrict__ zh){
  int i = blockIdx.x*256 + threadIdx.x;          // B*L
  if (i >= BB*LL) return;
  int b = i >> 6, l = i & 63;
  float mu = logz[b*2*LL + l], lv = logz[b*2*LL + LL + l];
  float z = mu + expf(0.5f*lv) * eps[i];
  sz_out[i] = z; zh[i] = (_Float16)z;
}
__global__ void mask_update_kernel(const float* __restrict__ sb, float* __restrict__ logmask,
                                   float* __restrict__ mask, float* __restrict__ masks_out){
  int b = blockIdx.x, t = threadIdx.x;
  __shared__ float sc[256];
  sc[t] = sb[b*TT + t]; __syncthreads();
  for (int off = 1; off < 256; off <<= 1){
    float v = (t >= off) ? sc[t - off] : 0.f;
    __syncthreads();
    sc[t] += v;
    __syncthreads();
  }
  float lm = logmask[b*TT + t] + logf(sc[t] + 1e-17f);
  logmask[b*TT + t] = lm;
  float m = expf(lm);
  mask[b*TT + t] = m;
  masks_out[b*TT + t] = m;
}
__global__ void bcast_rec_kernel(const float* __restrict__ pred, float* __restrict__ rec_seg){
  size_t i = (size_t)blockIdx.x*256 + threadIdx.x;   // B*T*V
  if (i >= (size_t)BB*TT*VV) return;
  int v = (int)(i % VV);
  int b = (int)((i / VV) / TT);
  rec_seg[i] = pred[(size_t)b*VV + v];
}

// ---------------- host driver ----------------
extern "C" void kernel_launch(void* const* d_in, const int* in_sizes, int n_in,
                              void* d_out, int out_size, void* d_ws, size_t ws_size,
                              hipStream_t stream)
{
  (void)in_sizes; (void)n_in; (void)out_size; (void)ws_size;
  const int*   actions = (const int*)  d_in[0];
  const int*   lengths = (const int*)  d_in[1];
  const float* gumbel  = (const float*)d_in[2];
  const float* eps_z   = (const float*)d_in[3];
  const float* embed   = (const float*)d_in[4];
  const float* Wx      = (const float*)d_in[5];
  const float* Wh      = (const float*)d_in[6];
  const float* b_lstm  = (const float*)d_in[7];
  const float* Wz1     = (const float*)d_in[8];
  const float* bz1     = (const float*)d_in[9];
  const float* Wz2     = (const float*)d_in[10];
  const float* bz2     = (const float*)d_in[11];
  const float* Wb1     = (const float*)d_in[12];
  const float* bb1     = (const float*)d_in[13];
  const float* Wb2     = (const float*)d_in[14];
  const float* bb2     = (const float*)d_in[15];
  const float* Wd1     = (const float*)d_in[16];
  const float* bd1     = (const float*)d_in[17];
  const float* Wd2     = (const float*)d_in[18];
  const float* bd2     = (const float*)d_in[19];
  float* out = (float*)d_out;
  char*  ws  = (char*)d_ws;

  _Float16* Wcat_h = (_Float16*)(ws + WS_WCAT);
  _Float16* X_h    = (_Float16*)(ws + WS_X);
  _Float16* ench   = (_Float16*)(ws + WS_ENCH);
  _Float16* Wz1_h  = (_Float16*)(ws + WS_WZ1);
  _Float16* Wz2_h  = (_Float16*)(ws + WS_WZ2);
  _Float16* Wb1_h  = (_Float16*)(ws + WS_WB1);
  _Float16* Wd1_h  = (_Float16*)(ws + WS_WD1);
  _Float16* Wd2p_h = (_Float16*)(ws + WS_WD2);
  _Float16* h0_h   = (_Float16*)(ws + WS_H0);
  _Float16* h1_h   = (_Float16*)(ws + WS_H1);
  float*    c_f    = (float*)   (ws + WS_C);
  float*    mask_f = (float*)   (ws + WS_MASK);
  float*    logm_f = (float*)   (ws + WS_LOGM);
  float*    lgw_f  = (float*)   (ws + WS_LGW);
  float*    sb_w   = (float*)   (ws + WS_SB);
  _Float16* rd_h   = (_Float16*)(ws + WS_RDH);
  _Float16* hz_h   = (_Float16*)(ws + WS_HZ);
  _Float16* z_h    = (_Float16*)(ws + WS_Z);
  _Float16* hd_h   = (_Float16*)(ws + WS_HD);
  float*    pred_f = (float*)   (ws + WS_PRED);

  // weight conversions (re-done each call for determinism)
  cvt_kernel<<<(HH*H4+255)/256,256,0,stream>>>(Wx, Wcat_h, (size_t)HH*H4);
  cvt_kernel<<<(HH*H4+255)/256,256,0,stream>>>(Wh, Wcat_h + (size_t)HH*H4, (size_t)HH*H4);
  cvt_kernel<<<(HH*HH+255)/256,256,0,stream>>>(Wz1, Wz1_h, (size_t)HH*HH);
  cvt_kernel<<<(HH*2*LL+255)/256,256,0,stream>>>(Wz2, Wz2_h, (size_t)HH*2*LL);
  cvt_kernel<<<(HH*HH+255)/256,256,0,stream>>>(Wb1, Wb1_h, (size_t)HH*HH);
  cvt_kernel<<<(LL*HH+255)/256,256,0,stream>>>(Wd1, Wd1_h, (size_t)LL*HH);
  cvt_pad_kernel<<<(512*512+255)/256,256,0,stream>>>(Wd2, Wd2p_h);
  gather_kernel<<<BT,256,0,stream>>>(actions, embed, X_h);
  init_mask_kernel<<<BT/256,256,0,stream>>>(mask_f, logm_f);

  for (int seg = 0; seg < SS; ++seg){
    init_hc_kernel<<<BB*HH/256,256,0,stream>>>(h0_h, h1_h, c_f);
    for (int t = 0; t < TT; ++t){
      _Float16* hr = (t & 1) ? h1_h : h0_h;
      _Float16* hw = (t & 1) ? h0_h : h1_h;
      lstm_step_kernel<<<dim3(HH/16, 2),256,0,stream>>>(X_h, Wcat_h, b_lstm, mask_f, hr, hw, c_f,
                                                        out + ENC_OFF, seg, t);
    }
    const float* enc_seg = out + ENC_OFF + (size_t)seg*BB*TT*HH;

    if (seg < SS-1){
      cvt_kernel<<<(int)(((size_t)BT*HH+255)/256),256,0,stream>>>(enc_seg, ench, (size_t)BT*HH);
      fill_scalar_kernel<<<BT/256,256,0,stream>>>(lgw_f, bb2, BT);
      gemm_wmma_kernel<<<dim3(HH/64, BT/64),256,0,stream>>>(ench, HH, Wb1_h, HH, bb1, HH, 2,
          nullptr, 0, HH, nullptr, 0, Wb2, lgw_f);
      softmax_gumbel_kernel<<<BB,256,0,stream>>>(lgw_f, gumbel + (size_t)seg*BT,
          out + LOGB_OFF + (size_t)seg*BT, out + SB_OFF + (size_t)seg*BT, sb_w);
    } else {
      onehot_kernel<<<BB,256,0,stream>>>(lengths, out + SB_OFF + (size_t)seg*BT, sb_w);
    }

    readout_kernel<<<BB*HH/256,256,0,stream>>>(enc_seg, sb_w, rd_h);
    gemm_wmma_kernel<<<dim3(HH/64,1),256,0,stream>>>(rd_h, HH, Wz1_h, HH, bz1, HH, 1,
        nullptr, 0, HH, hz_h, HH, nullptr, nullptr);
    gemm_wmma_kernel<<<dim3(2*LL/64,1),256,0,stream>>>(hz_h, HH, Wz2_h, 2*LL, bz2, HH, 0,
        out + LOGZ_OFF + (size_t)seg*BB*2*LL, 2*LL, 2*LL, nullptr, 0, nullptr, nullptr);
    samplez_kernel<<<(BB*LL+255)/256,256,0,stream>>>(out + LOGZ_OFF + (size_t)seg*BB*2*LL,
        eps_z + (size_t)seg*BB*LL, out + SZ_OFF + (size_t)seg*BB*LL, z_h);
    gemm_wmma_kernel<<<dim3(HH/64,1),256,0,stream>>>(z_h, LL, Wd1_h, HH, bd1, LL, 1,
        nullptr, 0, HH, hd_h, HH, nullptr, nullptr);
    gemm_wmma_kernel<<<dim3(512/64,1),256,0,stream>>>(hd_h, HH, Wd2p_h, 512, bd2, HH, 0,
        pred_f, VV, VV, nullptr, 0, nullptr, nullptr);
    bcast_rec_kernel<<<(int)(((size_t)BB*TT*VV+255)/256),256,0,stream>>>(
        pred_f, out + REC_OFF + (size_t)seg*BB*TT*VV);

    if (seg < SS-1){
      mask_update_kernel<<<BB,256,0,stream>>>(sb_w, logm_f, mask_f,
          out + MASK_OFF + (size_t)seg*BT);
    }
  }
}
